// DCNv2_4157528342738
// MI455X (gfx1250) — matrix operand order repaired and checked by reading, not compiled, not run on hardware
//
#include <hip/hip_runtime.h>

typedef __attribute__((ext_vector_type(16))) _Float16 v16h;
typedef __attribute__((ext_vector_type(8)))  _Float16 v8h;
typedef __attribute__((ext_vector_type(8)))  float    v8f;
typedef __attribute__((ext_vector_type(4)))  float    v4f;

#define BS_   4
#define IH_   64
#define IW_   64
#define IC_   256
#define F_    256
#define KS_   9
#define KDIM  (KS_*IC_)          // 2304
#define NPIX  (BS_*IH_*IW_)      // 16384
#define MTILE 32                 // pixels per workgroup (2 WMMA M-tiles)

// ---------------------------------------------------------------------------
// Kernel 1: convert conv weights fp32 [K=2304][N=256] -> f16 transposed
//           Wt[N][K] so WMMA B-fragments read contiguous K per column.
// ---------------------------------------------------------------------------
__global__ __launch_bounds__(256) void dcn_prep_w(const float* __restrict__ w,
                                                  _Float16* __restrict__ wt) {
  int idx = blockIdx.x * 256 + threadIdx.x;     // 0 .. K*F-1, write-coalesced
  int n = idx / KDIM;
  int k = idx - n * KDIM;
  wt[idx] = (_Float16)w[k * F_ + n];
}

// ---------------------------------------------------------------------------
// Kernel 2: offset branch 3x3 SAME conv -> off[pix][27]
//           j<18: raw (dy,dx per tap);  j>=18: sigmoid (modulation mask).
//           One wave per pixel; patch staged in LDS.
// ---------------------------------------------------------------------------
__global__ __launch_bounds__(32) void dcn_offset_conv(const float* __restrict__ x,
                                                      const float* __restrict__ ow,
                                                      const float* __restrict__ ob,
                                                      float* __restrict__ off) {
  __shared__ float patch[KDIM];                 // 9 KB
  int pix = blockIdx.x;
  int b = pix >> 12;
  int rem = pix & 4095;
  int y  = rem >> 6;
  int xc = rem & 63;
  int t = threadIdx.x;

  for (int i = t; i < KDIM; i += 32) {
    int tap = i >> 8, c = i & 255;
    int yy = y + tap / 3 - 1;
    int xx = xc + tap % 3 - 1;
    float v = 0.f;
    if (yy >= 0 && yy < IH_ && xx >= 0 && xx < IW_)
      v = x[(((b * IH_ + yy) * IW_ + xx) << 8) + c];
    patch[i] = v;
  }
  __syncthreads();

  if (t < 27) {
    float acc = ob[t];
#pragma unroll 8
    for (int kk = 0; kk < KDIM; ++kk)
      acc = fmaf(patch[kk], ow[kk * 27 + t], acc);
    if (t >= 18) acc = 1.f / (1.f + __expf(-acc));
    off[pix * 27 + t] = acc;
  }
}

// ---------------------------------------------------------------------------
// Kernel 3: fused deformable sampling + WMMA GEMM.
//   Workgroup = 256 threads (8 waves) -> 32 pixels x 256 filters.
//   Phase 1: build A tile (32 x 2304, f16) in LDS, rows interleaved
//            (slot = (p&15)*2 + (p>>4)) so both M-tile fragments sit within
//            the 16-bit DS immediate-offset range of one base address.
//   Phase 2: double-buffered k-loop; per k-step 2 LDS A-frags + 2 global
//            B-frags feed 4 WMMAs with no fragment-register reuse hazards.
// ---------------------------------------------------------------------------
__device__ __forceinline__ v4f load4z(const float* p, bool valid) {
  v4f z = {0.f, 0.f, 0.f, 0.f};
  return valid ? *(const v4f*)p : z;
}

__global__ __launch_bounds__(256) void dcn_main(const float* __restrict__ x,
                                                const _Float16* __restrict__ wt,
                                                const float* __restrict__ bias,
                                                const float* __restrict__ off,
                                                float* __restrict__ out) {
  extern __shared__ _Float16 lds_a[];           // 32 * 2304 halves = 144 KB
  const int tid  = threadIdx.x;
  const int lane = tid & 31;
  const int wave = tid >> 5;
  const int pix_base = blockIdx.x * MTILE;

  // ---------------- Phase 1: sample 288 (pixel,tap) pairs into LDS ---------
#pragma unroll 1
  for (int i = 0; i < 36; ++i) {
    int pair = wave * 36 + i;                   // uniform within the wave
    int p   = pair / 9;
    int tap = pair - p * 9;
    int pix = pix_base + p;
    int b = pix >> 12;
    int rem = pix & 4095;
    int y  = rem >> 6;
    int xc = rem & 63;

    const float* o = off + pix * 27;
    float dy = o[2 * tap];
    float dx = o[2 * tap + 1];
    float m  = o[18 + tap];

    // padded-coordinate sample position: g = (y + r + dy, x + s + dx)
    float gy = (float)(y + tap / 3) + dy;
    float gx = (float)(xc + tap % 3) + dx;

    const float B1 = 65.f;                      // clip bound ih+1 (padded coords)
    float y0f = floorf(gy);
    float y1c = fminf(fmaxf(y0f + 1.f, 0.f), B1);
    float y0c = fminf(fmaxf(y0f, 0.f), B1);
    float gyc = fminf(fmaxf(gy, 0.f), B1);
    float x0f = floorf(gx);
    float x1c = fminf(fmaxf(x0f + 1.f, 0.f), B1);
    float x0c = fminf(fmaxf(x0f, 0.f), B1);
    float gxc = fminf(fmaxf(gx, 0.f), B1);

    float dy0 = gyc - y0c, dy1 = y1c - gyc;
    float dx0 = gxc - x0c, dx1 = x1c - gxc;
    float w11 = m * dy0 * dx0;                  // sample (y1,x1)
    float w10 = m * dy0 * dx1;                  // sample (y1,x0)
    float w01 = m * dy1 * dx0;                  // sample (y0,x1)
    float w00 = m * dy1 * dx1;                  // sample (y0,x0)

    int y0i = (int)y0c, y1i = (int)y1c, x0i = (int)x0c, x1i = (int)x1c;
    bool vy0 = (y0i >= 1) && (y0i <= 64);
    bool vy1 = (y1i >= 1) && (y1i <= 64);
    bool vx0 = (x0i >= 1) && (x0i <= 64);
    bool vx1 = (x1i >= 1) && (x1i <= 64);

    const float* xb  = x + ((size_t)b << 20);   // b * 64*64*256
    const float* p00 = xb + ((((y0i - 1) * IW_) + (x0i - 1)) << 8);
    const float* p01 = xb + ((((y0i - 1) * IW_) + (x1i - 1)) << 8);
    const float* p10 = xb + ((((y1i - 1) * IW_) + (x0i - 1)) << 8);
    const float* p11 = xb + ((((y1i - 1) * IW_) + (x1i - 1)) << 8);

    int c0 = lane << 3;                         // 8 channels per lane
    v8h hv;
#pragma unroll
    for (int h = 0; h < 8; h += 4) {
      int c = c0 + h;
      v4f s00 = load4z(p00 + c, vy0 && vx0);
      v4f s01 = load4z(p01 + c, vy0 && vx1);
      v4f s10 = load4z(p10 + c, vy1 && vx0);
      v4f s11 = load4z(p11 + c, vy1 && vx1);
      v4f r = w11 * s11 + w10 * s10 + w01 * s01 + w00 * s00;
      hv[h + 0] = (_Float16)r.x;
      hv[h + 1] = (_Float16)r.y;
      hv[h + 2] = (_Float16)r.z;
      hv[h + 3] = (_Float16)r.w;
    }
    // interleaved row layout: rows r and r+16 are adjacent
    int slot = ((p & 15) << 1) + (p >> 4);
    *(v8h*)(lds_a + slot * KDIM + tap * 256 + c0) = hv;   // 16B aligned
  }

  __syncthreads();

  // ---------------- Phase 2: WMMA GEMM, 2 M-tiles x 2 N-tiles per wave -----
  const int col = lane & 15;
  // A fragment (16-bit 16x32): lanes 0-15 hold K 0..15, lanes 16-31 hold K 16..31
  const int ka = (lane < 16) ? 0 : 16;
  // B fragment (16-bit 32x16): lanes 0-15 hold K 0..15, lanes 16-31 hold K 16..31
  const int kb = (lane < 16) ? 0 : 16;
  const int n0 = wave * 32;                     // 8 waves x 32 filters = 256

  // interleaved rows: M-tile0 row `col` at slot 2*col, M-tile1 row `col` at 2*col+1
  const _Float16* a_base = lds_a + col * 2 * KDIM + ka;    // +KDIM for M-tile1
  const _Float16* bb0 = wt + (size_t)(n0 + col) * KDIM + kb;
  const _Float16* bb1 = bb0 + 16 * KDIM;

  v8f z8 = {0.f, 0.f, 0.f, 0.f, 0.f, 0.f, 0.f, 0.f};
  v8f acc00 = z8, acc01 = z8, acc10 = z8, acc11 = z8;  // acc[m][n]

  // prologue: fragments for k0 = 0
  v16h a0, a1, b0, b1;
  *(v8h*)&a0       = *(const v8h*)(a_base);
  *((v8h*)&a0 + 1) = *(const v8h*)(a_base + 8);
  *(v8h*)&a1       = *(const v8h*)(a_base + KDIM);
  *((v8h*)&a1 + 1) = *(const v8h*)(a_base + KDIM + 8);
  b0 = *(const v16h*)(bb0);
  b1 = *(const v16h*)(bb1);

#pragma unroll 2
  for (int k0 = 0; k0 < KDIM; k0 += 32) {
    v16h ca0 = a0, ca1 = a1, cb0 = b0, cb1 = b1;

    int kn = k0 + 32;
    if (kn >= KDIM) kn = 0;                     // harmless dummy reload on last iter
    *(v8h*)&a0       = *(const v8h*)(a_base + kn);
    *((v8h*)&a0 + 1) = *(const v8h*)(a_base + kn + 8);
    *(v8h*)&a1       = *(const v8h*)(a_base + KDIM + kn);
    *((v8h*)&a1 + 1) = *(const v8h*)(a_base + KDIM + kn + 8);
    b0 = *(const v16h*)(bb0 + kn);
    b1 = *(const v16h*)(bb1 + kn);

    acc00 = __builtin_amdgcn_wmma_f32_16x16x32_f16(false, ca0, false, cb0, (short)0, acc00, false, false);
    acc01 = __builtin_amdgcn_wmma_f32_16x16x32_f16(false, ca0, false, cb1, (short)0, acc01, false, false);
    acc10 = __builtin_amdgcn_wmma_f32_16x16x32_f16(false, ca1, false, cb0, (short)0, acc10, false, false);
    acc11 = __builtin_amdgcn_wmma_f32_16x16x32_f16(false, ca1, false, cb1, (short)0, acc11, false, false);
  }

  // C/D layout: VGPR r -> M = r + (lane>=16 ? 8 : 0), N = lane&15
  const int mbase = (lane < 16) ? 0 : 8;
  v8f accs[2][2] = {{acc00, acc01}, {acc10, acc11}};
#pragma unroll
  for (int mt = 0; mt < 2; ++mt) {
#pragma unroll
    for (int nt = 0; nt < 2; ++nt) {
      int n = n0 + nt * 16 + col;
      float bv = bias[n];
#pragma unroll
      for (int r = 0; r < 8; ++r) {
        int pix = pix_base + mt * 16 + mbase + r;
        out[(size_t)pix * F_ + n] = accs[mt][nt][r] + bv;
      }
    }
  }
}

// ---------------------------------------------------------------------------
extern "C" void kernel_launch(void* const* d_in, const int* in_sizes, int n_in,
                              void* d_out, int out_size, void* d_ws, size_t ws_size,
                              hipStream_t stream) {
  const float* x    = (const float*)d_in[0];   // [4,64,64,256]
  const float* w    = (const float*)d_in[1];   // [3,3,256,256]
  const float* bias = (const float*)d_in[2];   // [256]
  const float* ow   = (const float*)d_in[3];   // [3,3,256,27]
  const float* ob   = (const float*)d_in[4];   // [27]
  float* out = (float*)d_out;                  // [4,64,64,256]

  float*    off = (float*)d_ws;                                  // 16384*27 f32
  _Float16* wt  = (_Float16*)((char*)d_ws +
                              (size_t)NPIX * 27 * sizeof(float)); // 256*2304 f16

  dcn_prep_w<<<(KDIM * F_) / 256, 256, 0, stream>>>(w, wt);
  dcn_offset_conv<<<NPIX, 32, 0, stream>>>(x, ow, ob, off);
  dcn_main<<<NPIX / MTILE, 256, MTILE * KDIM * sizeof(_Float16), stream>>>(x, wt, bias, off, out);
}
